// MMTrackerOriginalCorrelation_69578470195530
// MI455X (gfx1250) — matrix operand order repaired and checked by reading, not compiled, not run on hardware
//
#include <hip/hip_runtime.h>

typedef _Float16 v8h  __attribute__((ext_vector_type(8)));
typedef _Float16 v16h __attribute__((ext_vector_type(16)));
typedef _Float16 h2   __attribute__((ext_vector_type(2)));
typedef float    v8f  __attribute__((ext_vector_type(8)));
typedef float    v4f  __attribute__((ext_vector_type(4)));

#define BB 16
#define CC 256
#define HH 64
#define WW 64
#define DD 4          // max displacement
#define PP 9          // patch size
#define CH_HALF 128   // channels staged per pass
#define XSTR 136      // halfs per pixel column in LDS (128 + 8 pad; 272B = 17*16B)
#define XCOLS 64
#define YCOLS 88      // cols -4..83 (cols beyond 67 zero-filled; D2 waste band reads stay in-bounds)
#define XHALFS (XCOLS * XSTR)                 // 8704
#define YHALFS (YCOLS * XSTR)                 // 11968
#define SCR_OFF ((XHALFS + YHALFS) * 2)       // 41344 bytes
#define SCR_STRIDE 20                         // floats per scratch row (80B, 16B aligned)
#define SCR_ROWS 24                           // band columns 0..23
#define LDS_BYTES (SCR_OFF + 4 * SCR_ROWS * SCR_STRIDE * 4)  // 41344 + 7680 = 49024

__global__ __launch_bounds__(128) void corr_wmma_kernel(
    const float* __restrict__ x, const float* __restrict__ y,
    float* __restrict__ out) {
  extern __shared__ char smem[];
  _Float16* xls = (_Float16*)smem;            // xls[j][c]  j:0..63
  _Float16* yls = xls + XHALFS;               // yls[jc][c] jc = j+4, 0..87

  const int tid  = threadIdx.x;
  const int lane = tid & 31;
  const int wv   = tid >> 5;                  // wave -> j-tile
  const int bid  = blockIdx.x;
  const int di   = bid % PP;
  const int rem  = bid / PP;
  const int i    = rem % HH;
  const int b    = rem / HH;
  const int ry   = i + di - DD;               // y source row (may be OOB -> zeros)
  const bool rv  = (ry >= 0) && (ry < HH);

  const int js = tid & 63;                    // staging column
  const int cg = tid >> 6;                    // staging channel group (0/1)
  const int j0 = wv * 16;
  const int nn = lane & 15;
  const int hi = lane >> 4;

  v8f acc1, acc2;
  #pragma unroll
  for (int r = 0; r < 8; ++r) { acc1[r] = 0.0f; acc2[r] = 0.0f; }

  // ---- zero-pad border columns once: jc 0..3 (j=-4..-1) and jc 68..87 (j>=64) ----
  // (never overwritten by restaging; ordered by the pass-entry barrier below)
  {
    h2 z; z[0] = (_Float16)0.0f; z[1] = (_Float16)0.0f;
    for (int t = tid; t < 24 * 64; t += 128) {
      const int colz = t >> 6;
      const int jc = (colz < 4) ? colz : (colz + 64);
      const int cl = (t & 63) * 2;
      *(h2*)&yls[jc * XSTR + cl] = z;
    }
  }

  for (int ch = 0; ch < 2; ++ch) {
    __syncthreads();
    const int cbase = ch * CH_HALF;

    // ---- stage x row i (f32 -> f16, transpose to channel-contiguous) ----
    const float* xp = x + (((size_t)(b * CC + cbase) * HH + i) * WW) + js;
    #pragma unroll 8
    for (int k = 0; k < 32; ++k) {
      const int cl = 4 * k + 2 * cg;
      const float f0 = xp[(size_t)cl * (HH * WW)];
      const float f1 = xp[(size_t)(cl + 1) * (HH * WW)];
      h2 v; v[0] = (_Float16)f0; v[1] = (_Float16)f1;
      *(h2*)&xls[js * XSTR + cl] = v;
    }

    // ---- stage y row i+di-4 (block-uniform branch hoisted out of loop) ----
    if (rv) {
      const float* yp = y + (((size_t)(b * CC + cbase) * HH + ry) * WW) + js;
      #pragma unroll 8
      for (int k = 0; k < 32; ++k) {
        const int cl = 4 * k + 2 * cg;
        const float f0 = yp[(size_t)cl * (HH * WW)];
        const float f1 = yp[(size_t)(cl + 1) * (HH * WW)];
        h2 v; v[0] = (_Float16)f0; v[1] = (_Float16)f1;
        *(h2*)&yls[(js + DD) * XSTR + cl] = v;
      }
    } else {
      h2 z; z[0] = (_Float16)0.0f; z[1] = (_Float16)0.0f;
      #pragma unroll 8
      for (int k = 0; k < 32; ++k) {
        const int cl = 4 * k + 2 * cg;
        *(h2*)&yls[(js + DD) * XSTR + cl] = z;
      }
    }
    __syncthreads();

    // ---- WMMA over this 128-channel half: 4 K-chunks x 2 tiles ----
    const int arow  = (j0 + nn) * XSTR;       // A row m=nn : x pixel j0+nn
    const int brow1 = (j0 + nn) * XSTR;       // B1 col n=nn: y col j0-4+nn -> jc j0+nn
    const int brow2 = brow1 + 16 * XSTR;      // B2 col n=nn: y col j0+12+nn
    #pragma unroll
    for (int kc = 0; kc < 4; ++kc) {
      const int kb = kc * 32;
      union { v16h v; v8h h[2]; } A, B1, B2;
      // A 16x32 f16, k order chosen here (same as B): lane m=L%16; halves t=0..15 -> K=hi*16+t
      A.h[0] = *(const v8h*)&xls[arow + kb + hi * 16];
      A.h[1] = *(const v8h*)&xls[arow + kb + hi * 16 + 8];
      // B 32x16 f16, k order chosen here (same as A): lane n=L%16; halves t=0..15 -> K=hi*16+t
      B1.h[0] = *(const v8h*)&yls[brow1 + kb + hi * 16];
      B1.h[1] = *(const v8h*)&yls[brow1 + kb + hi * 16 + 8];
      B2.h[0] = *(const v8h*)&yls[brow2 + kb + hi * 16];
      B2.h[1] = *(const v8h*)&yls[brow2 + kb + hi * 16 + 8];
      acc1 = __builtin_amdgcn_wmma_f32_16x16x32_f16(false, A.v, false, B1.v,
                                                    (short)0, acc1, false, false);
      acc2 = __builtin_amdgcn_wmma_f32_16x16x32_f16(false, A.v, false, B2.v,
                                                    (short)0, acc2, false, false);
    }
  }

  // ---- band extraction via per-wave LDS scratch (column-major: scr[n][m]) ----
  float* scr = (float*)(smem + SCR_OFF) + wv * (SCR_ROWS * SCR_STRIDE);
  // D layout: lane n+16*hi holds rows m = 8*hi + r in acc[r] -> contiguous m run
  *(v4f*)&scr[nn * SCR_STRIDE + hi * 8] =
      (v4f){acc1[0], acc1[1], acc1[2], acc1[3]};
  *(v4f*)&scr[nn * SCR_STRIDE + hi * 8 + 4] =
      (v4f){acc1[4], acc1[5], acc1[6], acc1[7]};
  if (nn < 8) {  // D2 columns 16..23 only (band max col = 15+8 = 23)
    *(v4f*)&scr[(16 + nn) * SCR_STRIDE + hi * 8] =
        (v4f){acc2[0], acc2[1], acc2[2], acc2[3]};
    *(v4f*)&scr[(16 + nn) * SCR_STRIDE + hi * 8 + 4] =
        (v4f){acc2[4], acc2[5], acc2[6], acc2[7]};
  }

  // gather out[dj][m] = Dcat[m, m+dj] / C ; coalesced 16-lane stores per dj
  #pragma unroll
  for (int t = 0; t < 5; ++t) {
    const int idx = t * 32 + lane;
    if (idx < PP * 16) {
      const int dj = idx >> 4;
      const int m  = idx & 15;
      const int col = m + dj;
      const float val = scr[col * SCR_STRIDE + m] * (1.0f / (float)CC);
      out[(((size_t)(b * (PP * PP)) + di * PP + dj) * HH + i) * WW + j0 + m] = val;
    }
  }
}

extern "C" void kernel_launch(void* const* d_in, const int* in_sizes, int n_in,
                              void* d_out, int out_size, void* d_ws, size_t ws_size,
                              hipStream_t stream) {
  (void)in_sizes; (void)n_in; (void)d_ws; (void)ws_size; (void)out_size;
  const float* x = (const float*)d_in[0];
  const float* y = (const float*)d_in[1];
  float* out = (float*)d_out;
  const int nblocks = BB * HH * PP;  // 9216 blocks: (b, i, di)
  corr_wmma_kernel<<<dim3(nblocks), dim3(128), LDS_BYTES, stream>>>(x, y, out);
}
